// SymbioticGATBranch_18210661335430
// MI455X (gfx1250) — matrix operand order repaired and checked by reading, not compiled, not run on hardware
//
#include <hip/hip_runtime.h>
#include <hip/hip_bf16.h>

typedef __attribute__((ext_vector_type(16))) _Float16 v16h;
typedef __attribute__((ext_vector_type(8)))  _Float16 v8h;
typedef __attribute__((ext_vector_type(8)))  float    v8f;

#define BATCH 8
#define NNODE 1024
#define INDIM 64
#define HID   96
#define OUTD  128
#define NEG_SLOPE 0.2f

__device__ __forceinline__ float lrelu_f(float v) { return v > 0.0f ? v : NEG_SLOPE * v; }
__device__ __forceinline__ float elu_f(float v)   { return v > 0.0f ? v : (__expf(v) - 1.0f); }
__device__ __forceinline__ v16h cat8(v8h a, v8h b) {
  return __builtin_shufflevector(a, b, 0,1,2,3,4,5,6,7,8,9,10,11,12,13,14,15);
}

// ---------------------------------------------------------------------------
// LayerNorm over IN_DIM=64 -> f16 [B*N, 64]
// ---------------------------------------------------------------------------
__global__ void ln_kernel(const float* __restrict__ x, const float* __restrict__ g,
                          const float* __restrict__ b, _Float16* __restrict__ xn) {
  int row = blockIdx.x;
  int t = threadIdx.x;            // 64 threads
  __shared__ float sbuf[64];
  float v = x[(size_t)row * INDIM + t];
  sbuf[t] = v; __syncthreads();
  for (int s = 32; s > 0; s >>= 1) { if (t < s) sbuf[t] += sbuf[t + s]; __syncthreads(); }
  float mu = sbuf[0] * (1.0f / INDIM);
  __syncthreads();
  float d = v - mu;
  sbuf[t] = d * d; __syncthreads();
  for (int s = 32; s > 0; s >>= 1) { if (t < s) sbuf[t] += sbuf[t + s]; __syncthreads(); }
  float var = sbuf[0] * (1.0f / INDIM);
  float r = rsqrtf(var + 1e-5f);
  xn[(size_t)row * INDIM + t] = (_Float16)(d * r * g[t] + b[t]);
}

// ---------------------------------------------------------------------------
// Weight transpose + f16 convert: WT[col*K + k] = W[k*Nc + col]
// ---------------------------------------------------------------------------
__global__ void wtrans_kernel(const float* __restrict__ W, _Float16* __restrict__ WT,
                              int K, int Nc) {
  int idx = blockIdx.x * blockDim.x + threadIdx.x;
  if (idx >= K * Nc) return;
  int k = idx / Nc, c = idx % Nc;
  WT[(size_t)c * K + k] = (_Float16)W[idx];
}

// ---------------------------------------------------------------------------
// WMMA GEMM: C = act(A_f16[M,K] @ WT^T + bias); one wave / 16x16 tile.
// All operand fragments are contiguous 16B/32B runs (b128 loads).
// OUT_ROW16: row-major f16;  OUT_T16: transposed f16  hT[(b*HID+col)*N + node]
// ---------------------------------------------------------------------------
template <int ACT, bool HAS_BIAS, bool OUT_ROW16, bool OUT_T16>
__global__ void gemm16(const _Float16* __restrict__ A, const _Float16* __restrict__ WT,
                       const float* __restrict__ bias, _Float16* __restrict__ Crow,
                       _Float16* __restrict__ Ct, int K, int Nc) {
  int lane = threadIdx.x & 31;
  int half = lane >> 4, l15 = lane & 15;
  int m0 = blockIdx.x * 16, n0 = blockIdx.y * 16;
  int col = n0 + l15;
  const _Float16* arow = A + (size_t)(m0 + l15) * K;
  const _Float16* wcol = WT + (size_t)col * K;
  v8f acc = {};
  for (int k0 = 0; k0 < K; k0 += 32) {
    v8h a0 = *(const v8h*)(arow + k0 + 16 * half);         // A: k = 16h+0..7
    v8h a1 = *(const v8h*)(arow + k0 + 16 * half + 8);     //    k = 16h+8..15
    v16h av = cat8(a0, a1);
    v16h bv = *(const v16h*)(wcol + k0 + 16 * half);       // B: k = 16h+0..15
    acc = __builtin_amdgcn_wmma_f32_16x16x32_f16(false, av, false, bv, (short)0, acc,
                                                 false, false);
  }
  float bb = HAS_BIAS ? bias[col] : 0.0f;
#pragma unroll
  for (int r = 0; r < 8; ++r) {
    int row = m0 + r + 8 * half;
    float v = acc[r] + bb;
    if (ACT) v = elu_f(v);
    _Float16 hv = (_Float16)v;
    if (OUT_ROW16) Crow[(size_t)row * Nc + col] = hv;
    if (OUT_T16) {
      int bb2 = row >> 10, node = row & (NNODE - 1);
      Ct[((size_t)bb2 * HID + col) * NNODE + node] = hv;
    }
  }
}

// ---------------------------------------------------------------------------
// e_src/e_dst per (b,head,node) from transposed h
// ---------------------------------------------------------------------------
__global__ void edot_kernel(const _Float16* __restrict__ hT, const float* __restrict__ a_src,
                            const float* __restrict__ a_dst, float* __restrict__ e_src,
                            float* __restrict__ e_dst, int H, int dim) {
  int idx = blockIdx.x * blockDim.x + threadIdx.x;
  if (idx >= BATCH * H * NNODE) return;
  int n = idx % NNODE;
  int head = (idx / NNODE) % H;
  int b = idx / (NNODE * H);
  const _Float16* hcol = hT + (size_t)(b * HID + head * dim) * NNODE + n;
  float ss = 0.0f, sd = 0.0f;
  for (int d = 0; d < dim; ++d) {
    float hv = (float)hcol[(size_t)d * NNODE];
    ss += hv * a_src[head * dim + d];
    sd += hv * a_dst[head * dim + d];
  }
  e_src[idx] = ss;
  e_dst[idx] = sd;
}

// ---------------------------------------------------------------------------
// Fused GAT attention with WMMA.
//  - softmax stats per 16-row strip (cooperative)
//  - attention tile (16x32) computed once per block into LDS (double-buffered)
//  - hT tile (dim x 32) async-DMA'd into LDS (global_load_async_to_lds_b128)
//  - each wave owns one 16-col tile: ds_load A/B fragments -> v_wmma
// grid: (N/16, H, B); block: 2*dim threads
// ---------------------------------------------------------------------------
template <bool W16>
__global__ void attn_kernel(const _Float16* __restrict__ hT, const float* __restrict__ e_src,
                            const float* __restrict__ e_dst, const int* __restrict__ adj,
                            float* __restrict__ out32, _Float16* __restrict__ out16,
                            int H, int dim) {
  __shared__ float red[16 * 16];
  __shared__ float s_m[16], s_inv[16], s_srow[16];
  __shared__ __align__(16) _Float16 s_att[2][16 * 32];
  __shared__ __align__(16) _Float16 s_hb[2][HID * 32];

  int rows0 = blockIdx.x * 16;
  int head = blockIdx.y, b = blockIdx.z;
  const float* es = e_src + (size_t)(b * H + head) * NNODE;
  const float* ed = e_dst + (size_t)(b * H + head) * NNODE;

  // ---- per-row softmax stats ----
  int tid = threadIdx.x;
  int r = tid & 15, grp = tid >> 4, ngrp = blockDim.x >> 4;
  float s = es[rows0 + r];
  const int* arow = adj + (size_t)(rows0 + r) * NNODE;
  float mx = -3.0e38f;
  for (int j = grp; j < NNODE; j += ngrp) {
    float e = (arow[j] != 0) ? lrelu_f(s + ed[j]) : -1e9f;
    mx = fmaxf(mx, e);
  }
  red[r * 16 + grp] = mx; __syncthreads();
  if (grp == 0) {
    float m = red[r * 16];
    for (int g2 = 1; g2 < ngrp; ++g2) m = fmaxf(m, red[r * 16 + g2]);
    s_m[r] = m; s_srow[r] = s;
  }
  __syncthreads();
  float mi = s_m[r];
  float sum = 0.0f;
  for (int j = grp; j < NNODE; j += ngrp) {
    float e = (arow[j] != 0) ? lrelu_f(s + ed[j]) : -1e9f;
    sum += __expf(e - mi);
  }
  red[r * 16 + grp] = sum; __syncthreads();
  if (grp == 0) {
    float t = 0.0f;
    for (int g2 = 0; g2 < ngrp; ++g2) t += red[r * 16 + g2];
    s_inv[r] = 1.0f / t;
  }
  __syncthreads();

  // ---- main loop ----
  int lane = tid & 31, wave = tid >> 5;
  int half = lane >> 4, l15 = lane & 15;
  int colloc = wave * 16 + l15;                     // column within this head
  const _Float16* hbase = hT + (size_t)(b * HID + head * dim) * NNODE;
  int nchunk = dim * 4;                             // 16B async chunks per step
  v8f acc = {};
  for (int j0 = 0, it = 0; j0 < NNODE; j0 += 32, ++it) {
    int buf = it & 1;
    // async DMA: hT[dim cols x 32 j] -> LDS (overlaps with exp below)
    for (int c = tid; c < nchunk; c += blockDim.x) {
      int colc = c >> 2, part = c & 3;
      const _Float16* gsrc = hbase + (size_t)colc * NNODE + j0 + part * 8;
      unsigned lds = (unsigned)(uintptr_t)&s_hb[buf][colc * 32 + part * 8];
      asm volatile("global_load_async_to_lds_b128 %0, %1, off"
                   :: "v"(lds), "v"((unsigned long long)(uintptr_t)gsrc)
                   : "memory");
    }
    // attention tile (shared by all waves), computed once
    for (int idx = tid; idx < 512; idx += blockDim.x) {
      int i = idx >> 5, k = idx & 31, j = j0 + k;
      float ev = (adj[(size_t)(rows0 + i) * NNODE + j] != 0)
                     ? lrelu_f(s_srow[i] + ed[j]) : -1e9f;
      s_att[buf][idx] = (_Float16)(__expf(ev - s_m[i]) * s_inv[i]);
    }
    asm volatile("s_wait_asynccnt 0x0" ::: "memory");
    __syncthreads();
    v8h a0 = *(const v8h*)&s_att[buf][l15 * 32 + 16 * half];
    v8h a1 = *(const v8h*)&s_att[buf][l15 * 32 + 16 * half + 8];
    v16h av = cat8(a0, a1);
    v16h bv = *(const v16h*)&s_hb[buf][colloc * 32 + 16 * half];
    acc = __builtin_amdgcn_wmma_f32_16x16x32_f16(false, av, false, bv, (short)0, acc,
                                                 false, false);
  }
#pragma unroll
  for (int rr = 0; rr < 8; ++rr) {
    int mm = rr + 8 * half;
    float v = elu_f(acc[rr]);                       // ELU applied after every GAT here
    size_t idx = (size_t)(b * NNODE + rows0 + mm) * HID + head * dim + colloc;
    out32[idx] = v;
    if (W16) out16[idx] = (_Float16)v;
  }
}

// ---------------------------------------------------------------------------
// pooling / gate MLP / blends / projection (small scalar kernels)
// ---------------------------------------------------------------------------
__global__ void pool2_kernel(const float* __restrict__ x1, const float* __restrict__ x2,
                             float* __restrict__ pooled) {
  int idx = blockIdx.x * blockDim.x + threadIdx.x;
  if (idx >= BATCH * 2 * HID) return;
  int f = idx % (2 * HID);
  int b = idx / (2 * HID);
  const float* src = (f < HID) ? x1 : x2;
  int c = f % HID;
  float s = 0.0f;
  for (int n = 0; n < NNODE; ++n) s += src[(size_t)(b * NNODE + n) * HID + c];
  pooled[idx] = s * (1.0f / NNODE);
}

__global__ void gate_kernel(const float* __restrict__ pooled, const float* __restrict__ w1,
                            const float* __restrict__ b1, const float* __restrict__ w2,
                            const float* __restrict__ b2, float* __restrict__ gates,
                            float* __restrict__ dout) {
  __shared__ float hid[HID / 2];
  int b = blockIdx.x, t = threadIdx.x;    // 48 threads
  float acc = b1[t];
  for (int k = 0; k < 2 * HID; ++k) acc += pooled[b * 2 * HID + k] * w1[k * (HID / 2) + t];
  hid[t] = elu_f(acc);
  __syncthreads();
  if (t < 2) {
    float a = b2[t];
    for (int k = 0; k < HID / 2; ++k) a += hid[k] * w2[k * 2 + t];
    float g = 1.0f / (1.0f + __expf(-a));
    gates[b * 2 + t] = g;
    dout[BATCH * OUTD + t * BATCH + b] = g;   // gates[:,0] then gates[:,1]
  }
}

__global__ void blend_tdra_kernel(const float* __restrict__ x1, const float* __restrict__ x2,
                                  const float* __restrict__ gates,
                                  _Float16* __restrict__ td, _Float16* __restrict__ ra) {
  int idx = blockIdx.x * blockDim.x + threadIdx.x;
  if (idx >= BATCH * NNODE * HID) return;
  int b = idx / (NNODE * HID);
  float wt = gates[b * 2 + 0], wr = gates[b * 2 + 1];
  float a = x1[idx], c = x2[idx];
  td[idx] = (_Float16)((1.0f - wt) * a + wt * c);
  ra[idx] = (_Float16)((1.0f - wr) * a + wr * c);
}

__global__ void blend_stage_kernel(const float* __restrict__ x2, const float* __restrict__ x4,
                                   const float* __restrict__ x5, const float* __restrict__ gates,
                                   _Float16* __restrict__ stage) {
  int idx = blockIdx.x * blockDim.x + threadIdx.x;
  if (idx >= BATCH * NNODE * HID) return;
  int b = idx / (NNODE * HID);
  float wt = gates[b * 2 + 0], wr = gates[b * 2 + 1];
  stage[idx] = (_Float16)(x2[idx] + 0.5f * (wt * x4[idx] + wr * x5[idx]));
}

__global__ void mean3_kernel(const float* __restrict__ x3, const float* __restrict__ x4,
                             const float* __restrict__ x5, float* __restrict__ gt) {
  int idx = blockIdx.x * blockDim.x + threadIdx.x;
  if (idx >= BATCH * 3 * HID) return;
  int f = idx % (3 * HID);
  int b = idx / (3 * HID);
  const float* src = (f < HID) ? x3 : (f < 2 * HID ? x4 : x5);
  int c = f % HID;
  float s = 0.0f;
  for (int n = 0; n < NNODE; ++n) s += src[(size_t)(b * NNODE + n) * HID + c];
  gt[idx] = s * (1.0f / NNODE);
}

__global__ void proj_kernel(const float* __restrict__ gt, const float* __restrict__ w,
                            const float* __restrict__ bias, float* __restrict__ dout) {
  int b = blockIdx.x, o = threadIdx.x;    // 128 threads
  float acc = bias[o];
  for (int k = 0; k < 3 * HID; ++k) acc += gt[b * 3 * HID + k] * w[k * OUTD + o];
  dout[b * OUTD + o] = elu_f(acc);
}

// ---------------------------------------------------------------------------
// host
// ---------------------------------------------------------------------------
static char* carve(void* ws, size_t& off, size_t bytes) {
  char* p = (char*)ws + off;
  off += (bytes + 255) & ~(size_t)255;
  return p;
}

extern "C" void kernel_launch(void* const* d_in, const int* in_sizes, int n_in,
                              void* d_out, int out_size, void* d_ws, size_t ws_size,
                              hipStream_t stream) {
  const float* x     = (const float*)d_in[0];
  const int*   adj   = (const int*)d_in[1];
  const float* ln_g  = (const float*)d_in[2];
  const float* ln_b  = (const float*)d_in[3];
  const float* pre_w = (const float*)d_in[4];
  const float* pre_b = (const float*)d_in[5];
  const float* gw[5]  = {(const float*)d_in[6],  (const float*)d_in[9],  (const float*)d_in[12],
                         (const float*)d_in[15], (const float*)d_in[18]};
  const float* gas[5] = {(const float*)d_in[7],  (const float*)d_in[10], (const float*)d_in[13],
                         (const float*)d_in[16], (const float*)d_in[19]};
  const float* gad[5] = {(const float*)d_in[8],  (const float*)d_in[11], (const float*)d_in[14],
                         (const float*)d_in[17], (const float*)d_in[20]};
  const float* gate_w1 = (const float*)d_in[21];
  const float* gate_b1 = (const float*)d_in[22];
  const float* gate_w2 = (const float*)d_in[23];
  const float* gate_b2 = (const float*)d_in[24];
  const float* proj_w  = (const float*)d_in[25];
  const float* proj_b  = (const float*)d_in[26];
  float* out = (float*)d_out;

  const int M = BATCH * NNODE;              // 8192
  size_t off = 0;
  _Float16* xn16   = (_Float16*)carve(d_ws, off, (size_t)M * INDIM * 2);
  _Float16* x0_16  = (_Float16*)carve(d_ws, off, (size_t)M * HID * 2);
  _Float16* hT16   = (_Float16*)carve(d_ws, off, (size_t)M * HID * 2);   // transposed h
  float*    x1_32  = (float*)   carve(d_ws, off, (size_t)M * HID * 4);
  _Float16* x1_16  = (_Float16*)carve(d_ws, off, (size_t)M * HID * 2);
  float*    x2_32  = (float*)   carve(d_ws, off, (size_t)M * HID * 4);
  _Float16* td16   = (_Float16*)carve(d_ws, off, (size_t)M * HID * 2);
  _Float16* ra16   = (_Float16*)carve(d_ws, off, (size_t)M * HID * 2);
  float*    x4_32  = (float*)   carve(d_ws, off, (size_t)M * HID * 4);
  float*    x5_32  = (float*)   carve(d_ws, off, (size_t)M * HID * 4);
  _Float16* st16   = (_Float16*)carve(d_ws, off, (size_t)M * HID * 2);
  float*    x3_32  = (float*)   carve(d_ws, off, (size_t)M * HID * 4);
  float*    e_src  = (float*)   carve(d_ws, off, (size_t)3 * BATCH * NNODE * 4);
  float*    e_dst  = (float*)   carve(d_ws, off, (size_t)3 * BATCH * NNODE * 4);
  float*    pooled = (float*)   carve(d_ws, off, (size_t)BATCH * 2 * HID * 4);
  float*    gates  = (float*)   carve(d_ws, off, (size_t)BATCH * 2 * 4);
  float*    gt     = (float*)   carve(d_ws, off, (size_t)BATCH * 3 * HID * 4);
  _Float16* wt_pre = (_Float16*)carve(d_ws, off, (size_t)INDIM * HID * 2);
  _Float16* wt_g[5];
  for (int i = 0; i < 5; ++i) wt_g[i] = (_Float16*)carve(d_ws, off, (size_t)HID * HID * 2);

  int ew = BATCH * NNODE * HID;
  dim3 ggrid(M / 16, HID / 16);

  // weight transposes (f32 -> f16, column-major)
  wtrans_kernel<<<(INDIM * HID + 255) / 256, 256, 0, stream>>>(pre_w, wt_pre, INDIM, HID);
  for (int i = 0; i < 5; ++i)
    wtrans_kernel<<<(HID * HID + 255) / 256, 256, 0, stream>>>(gw[i], wt_g[i], HID, HID);

  // pre: LayerNorm -> Linear(64->96) + ELU
  ln_kernel<<<M, 64, 0, stream>>>(x, ln_g, ln_b, xn16);
  gemm16<1, true, true, false><<<ggrid, 32, 0, stream>>>(
      xn16, wt_pre, pre_b, x0_16, nullptr, INDIM, HID);

  // gat1: 3 heads x 32, concat
  gemm16<0, false, false, true><<<ggrid, 32, 0, stream>>>(
      x0_16, wt_g[0], nullptr, nullptr, hT16, HID, HID);
  edot_kernel<<<(BATCH * 3 * NNODE + 255) / 256, 256, 0, stream>>>(
      hT16, gas[0], gad[0], e_src, e_dst, 3, 32);
  attn_kernel<true><<<dim3(NNODE / 16, 3, BATCH), 64, 0, stream>>>(
      hT16, e_src, e_dst, adj, x1_32, x1_16, 3, 32);

  // gat2: 1 head x 96
  gemm16<0, false, false, true><<<ggrid, 32, 0, stream>>>(
      x1_16, wt_g[1], nullptr, nullptr, hT16, HID, HID);
  edot_kernel<<<(BATCH * NNODE + 255) / 256, 256, 0, stream>>>(
      hT16, gas[1], gad[1], e_src, e_dst, 1, 96);
  attn_kernel<false><<<dim3(NNODE / 16, 1, BATCH), 192, 0, stream>>>(
      hT16, e_src, e_dst, adj, x2_32, nullptr, 1, 96);

  // gates
  pool2_kernel<<<(BATCH * 2 * HID + 255) / 256, 256, 0, stream>>>(x1_32, x2_32, pooled);
  gate_kernel<<<BATCH, HID / 2, 0, stream>>>(pooled, gate_w1, gate_b1, gate_w2, gate_b2,
                                             gates, out);
  blend_tdra_kernel<<<(ew + 255) / 256, 256, 0, stream>>>(x1_32, x2_32, gates, td16, ra16);

  // gat4 on td_in
  gemm16<0, false, false, true><<<ggrid, 32, 0, stream>>>(
      td16, wt_g[3], nullptr, nullptr, hT16, HID, HID);
  edot_kernel<<<(BATCH * NNODE + 255) / 256, 256, 0, stream>>>(
      hT16, gas[3], gad[3], e_src, e_dst, 1, 96);
  attn_kernel<false><<<dim3(NNODE / 16, 1, BATCH), 192, 0, stream>>>(
      hT16, e_src, e_dst, adj, x4_32, nullptr, 1, 96);

  // gat5 on ra_in
  gemm16<0, false, false, true><<<ggrid, 32, 0, stream>>>(
      ra16, wt_g[4], nullptr, nullptr, hT16, HID, HID);
  edot_kernel<<<(BATCH * NNODE + 255) / 256, 256, 0, stream>>>(
      hT16, gas[4], gad[4], e_src, e_dst, 1, 96);
  attn_kernel<false><<<dim3(NNODE / 16, 1, BATCH), 192, 0, stream>>>(
      hT16, e_src, e_dst, adj, x5_32, nullptr, 1, 96);

  // gat3 on stage_in
  blend_stage_kernel<<<(ew + 255) / 256, 256, 0, stream>>>(x2_32, x4_32, x5_32, gates, st16);
  gemm16<0, false, false, true><<<ggrid, 32, 0, stream>>>(
      st16, wt_g[2], nullptr, nullptr, hT16, HID, HID);
  edot_kernel<<<(BATCH * NNODE + 255) / 256, 256, 0, stream>>>(
      hT16, gas[2], gad[2], e_src, e_dst, 1, 96);
  attn_kernel<false><<<dim3(NNODE / 16, 1, BATCH), 192, 0, stream>>>(
      hT16, e_src, e_dst, adj, x3_32, nullptr, 1, 96);

  // graph token + projection
  mean3_kernel<<<(BATCH * 3 * HID + 255) / 256, 256, 0, stream>>>(x3_32, x4_32, x5_32, gt);
  proj_kernel<<<BATCH, OUTD, 0, stream>>>(gt, proj_w, proj_b, out);
}
